// DyConv_86552180949428
// MI455X (gfx1250) — matrix operand order repaired and checked by reading, not compiled, not run on hardware
//
#include <hip/hip_runtime.h>
#include <hip/hip_bf16.h>

typedef __attribute__((ext_vector_type(16))) __bf16 v16bf;
typedef __attribute__((ext_vector_type(8)))  float  v8f;

union Frag16 {
    v16bf v;
    unsigned short u[16];
    uint4 q[2];
};

__device__ __forceinline__ unsigned short f2bf(float f) {
    unsigned int u = __float_as_uint(f);
    unsigned int r = (u + 0x7FFFu + ((u >> 16) & 1u)) >> 16;
    return (unsigned short)r;
}

__device__ __forceinline__ float hsig(float x) {
    return fminf(fmaxf(x + 3.0f, 0.0f), 6.0f) * (1.0f / 6.0f);
}

__device__ __forceinline__ float sigmoidf_(float x) {
    return 1.0f / (1.0f + __expf(-x));
}

// ---------------------------------------------------------------------------
// Pre-shuffle fp32 weights (Cout=256, Cin=256, 3, 3) into bf16 WMMA B-fragment
// order: [kc(72)][couttile(16)][lane(32)][j(16)], K = tap*256 + cin.
// ---------------------------------------------------------------------------
__global__ void wfrag_kernel(const float* __restrict__ w, unsigned short* __restrict__ out) {
    int idx = blockIdx.x * 256 + threadIdx.x;
    if (idx >= 72 * 16 * 32 * 16) return;
    int j    = idx & 15;
    int lane = (idx >> 4) & 31;
    int ct   = (idx >> 9) & 15;
    int kc   = idx >> 13;
    int K    = kc * 32 + ((lane >> 4) * 16) + j;   // B 32x16: lanes 0-15 K=0..15, 16-31 K=16..31
    int tap  = K >> 8;
    int cin  = K & 255;
    int cout = ct * 16 + (lane & 15);
    out[idx] = f2bf(w[(size_t)cout * 2304 + cin * 9 + tap]);
}

// ---------------------------------------------------------------------------
// Offset conv: 3x3, pad 1, stride 1, 256 -> 27 channels.
// ---------------------------------------------------------------------------
__global__ void offconv_kernel(const float* __restrict__ x, const float* __restrict__ ow,
                               const float* __restrict__ ob, float* __restrict__ om,
                               int H, int W) {
    int idx = blockIdx.x * 256 + threadIdx.x;
    int HW = H * W;
    int total = 4 * 27 * HW;
    if (idx >= total) return;
    int b  = idx / (27 * HW);
    int r0 = idx - b * 27 * HW;
    int oc = r0 / HW;
    int sp = r0 - oc * HW;
    int oy = sp / W, ox = sp - oy * W;
    const float* xb = x + (size_t)b * 256 * HW;
    const float* wr = ow + (size_t)oc * 2304;
    float acc = ob[oc];
    for (int c = 0; c < 256; ++c) {
        const float* xc = xb + (size_t)c * HW;
        const float* wc = wr + c * 9;
#pragma unroll
        for (int t = 0; t < 9; ++t) {
            int yy = oy - 1 + t / 3;
            int xx = ox - 1 + t % 3;
            if (yy >= 0 && yy < H && xx >= 0 && xx < W) acc += wc[t] * xc[yy * W + xx];
        }
    }
    om[idx] = acc;
}

// ---------------------------------------------------------------------------
// Fused deformable conv as implicit GEMM with bf16 WMMA.
// Block = 256 threads (8 waves): 16 pixels x 256 Cout.
// Loop nest: tap (9, descriptors hoisted) x chunk (8, fully unrolled).
// Branch-free clamped bilinear gather, double-buffered LDS A tile,
// software-pipelined B-fragment loads.
// ---------------------------------------------------------------------------
__global__ __launch_bounds__(256) void deform_wmma_kernel(
    const float* __restrict__ x, int Hi, int Wi,
    const float* __restrict__ om, int Hf, int Wf,
    int Ho, int Wo, int stride,
    const unsigned short* __restrict__ wf,
    const float* __restrict__ bias,
    float* __restrict__ out) {

    __shared__ float4         sW[16][9];      // mask-folded bilinear weights (0 when OOB)
    __shared__ int4           sO[16][9];      // absolute elem offsets (batch base folded in)
    __shared__ unsigned short sA[2][16][32];  // double-buffered bf16 A tile

    const int tid  = threadIdx.x;
    const int lane = tid & 31;
    const int wave = tid >> 5;
    const int HoWo = Ho * Wo;
    const int HiWi = Hi * Wi;
    const int baseP = blockIdx.x * 16;

    // Precompute per (pixel, tap): 4 clamped corner offsets + mask-folded weights.
    if (tid < 144) {
        int p = tid / 9, tap = tid - (tid / 9) * 9;
        int gp = baseP + p;
        int b = gp / HoWo;
        int rem = gp - b * HoWo;
        int oy = rem / Wo, ox = rem - oy * Wo;
        int bbase = b * 256 * HiWi;
        const float* omb = om + (size_t)b * 27 * Hf * Wf;
        int sp = oy * Wf + ox;
        float offy = omb[(tap * 2 + 0) * Hf * Wf + sp];
        float offx = omb[(tap * 2 + 1) * Hf * Wf + sp];
        float msk  = sigmoidf_(omb[(18 + tap) * Hf * Wf + sp]);
        int kh = tap / 3, kw = tap - kh * 3;
        float py = (float)(oy * stride - 1 + kh) + offy;
        float px = (float)(ox * stride - 1 + kw) + offx;
        float y0f = floorf(py), x0f = floorf(px);
        int iy0 = (int)y0f, ix0 = (int)x0f;
        float wy1 = py - y0f, wx1 = px - x0f;
        float wy0 = 1.0f - wy1, wx0 = 1.0f - wx1;
        float cw[4] = { wy0 * wx0 * msk, wy0 * wx1 * msk, wy1 * wx0 * msk, wy1 * wx1 * msk };
        int cy[4] = { iy0, iy0, iy0 + 1, iy0 + 1 };
        int cx[4] = { ix0, ix0 + 1, ix0, ix0 + 1 };
        float w4[4];
        int   o4[4];
#pragma unroll
        for (int c = 0; c < 4; ++c) {
            bool valid = (cy[c] >= 0) && (cy[c] < Hi) && (cx[c] >= 0) && (cx[c] < Wi);
            int yc = min(max(cy[c], 0), Hi - 1);
            int xc = min(max(cx[c], 0), Wi - 1);
            o4[c] = bbase + yc * Wi + xc;        // always a legal address
            w4[c] = valid ? cw[c] : 0.0f;        // zero weight kills OOB taps
        }
        sW[p][tap] = make_float4(w4[0], w4[1], w4[2], w4[3]);
        sO[p][tap] = make_int4(o4[0], o4[1], o4[2], o4[3]);
    }
    __syncthreads();

    v8f acc0 = {};
    v8f acc1 = {};
    const int ct0 = wave * 2, ct1 = wave * 2 + 1;
    const int kk = tid & 31;       // K element within chunk
    const int p0 = tid >> 5;       // pixels p0 and p0+8
    const int m  = lane & 15, kb = (lane >> 4) * 16;

    // Software-pipelined B fragments: preload chunk 0.
    Frag16 B0c, B1c;
    {
        const uint4* q0 = (const uint4*)(wf + ((size_t)ct0 * 32 + lane) * 16);
        const uint4* q1 = (const uint4*)(wf + ((size_t)ct1 * 32 + lane) * 16);
        B0c.q[0] = q0[0]; B0c.q[1] = q0[1];
        B1c.q[0] = q1[0]; B1c.q[1] = q1[1];
    }

    for (int tap = 0; tap < 9; ++tap) {
        // Descriptors are tap-invariant across the 8 chunks: hoist.
        const int4   oA = sO[p0][tap];
        const float4 wA = sW[p0][tap];
        const int4   oB = sO[p0 + 8][tap];
        const float4 wB = sW[p0 + 8][tap];

#pragma unroll
        for (int q = 0; q < 8; ++q) {
            const int kc  = tap * 8 + q;
            const int buf = q & 1;

            // Branch-free bilinear gather: 2 im2col elements per thread (32-bit addressing).
            const float* xc = x + (q * 32 + kk) * HiWi;
            float a0 = wA.x * xc[oA.x] + wA.y * xc[oA.y] + wA.z * xc[oA.z] + wA.w * xc[oA.w];
            float a1 = wB.x * xc[oB.x] + wB.y * xc[oB.y] + wB.z * xc[oB.z] + wB.w * xc[oB.w];
            sA[buf][p0][kk]     = f2bf(a0);
            sA[buf][p0 + 8][kk] = f2bf(a1);
            __syncthreads();   // single barrier per step (ping-pong buffers)

            // A fragment (16x32 bf16): lanes 0-15 K=0..15, lanes 16-31 K=16..31.
            Frag16 A;
#pragma unroll
            for (int j = 0; j < 8; ++j) {
                A.u[j]     = sA[buf][m][kb + j];
                A.u[8 + j] = sA[buf][m][kb + 8 + j];
            }

            // Preload next chunk's B fragments so WMMA never waits on fresh loads.
            const int kn = (kc < 71) ? (kc + 1) : 71;
            Frag16 B0n, B1n;
            {
                const uint4* q0 = (const uint4*)(wf + (((size_t)kn * 16 + ct0) * 32 + lane) * 16);
                const uint4* q1 = (const uint4*)(wf + (((size_t)kn * 16 + ct1) * 32 + lane) * 16);
                B0n.q[0] = q0[0]; B0n.q[1] = q0[1];
                B1n.q[0] = q1[0]; B1n.q[1] = q1[1];
            }

            acc0 = __builtin_amdgcn_wmma_f32_16x16x32_bf16(false, A.v, false, B0c.v,
                                                           (short)0, acc0, false, false);
            acc1 = __builtin_amdgcn_wmma_f32_16x16x32_bf16(false, A.v, false, B1c.v,
                                                           (short)0, acc1, false, false);
            B0c = B0n;
            B1c = B1n;
        }
    }

    // D 16x16 f32: lane col = lane&15, rows M = (lane>>4)*8 + r.
    const int col = lane & 15, hb = lane >> 4;
    const int co0 = ct0 * 16 + col, co1 = ct1 * 16 + col;
    const float bi0 = bias[co0], bi1 = bias[co1];
#pragma unroll
    for (int r = 0; r < 8; ++r) {
        int p  = hb * 8 + r;
        int gp = baseP + p;
        int b  = gp / HoWo;
        int rem = gp - b * HoWo;
        out[((size_t)(b * 256 + co0)) * HoWo + rem] = acc0[r] + bi0;
        out[((size_t)(b * 256 + co1)) * HoWo + rem] = acc1[r] + bi1;
    }
}

// ---------------------------------------------------------------------------
// GroupNorm (groups=16, 16 ch/group, contiguous): stats then in-place apply.
// ---------------------------------------------------------------------------
__global__ void gn_stats_kernel(const float* __restrict__ v, int HW, float* __restrict__ stat) {
    __shared__ float ss[256], sq[256];
    int bg = blockIdx.x;                 // b*16 + g
    int b = bg >> 4, g = bg & 15;
    const float* base = v + ((size_t)b * 256 + g * 16) * HW;
    int n = 16 * HW;
    float s = 0.0f, q = 0.0f;
    for (int i = threadIdx.x; i < n; i += 256) { float t = base[i]; s += t; q += t * t; }
    ss[threadIdx.x] = s; sq[threadIdx.x] = q;
    __syncthreads();
    for (int st = 128; st > 0; st >>= 1) {
        if (threadIdx.x < st) { ss[threadIdx.x] += ss[threadIdx.x + st]; sq[threadIdx.x] += sq[threadIdx.x + st]; }
        __syncthreads();
    }
    if (threadIdx.x == 0) {
        float m = ss[0] / n;
        float var = sq[0] / n - m * m;
        stat[bg * 2]     = m;
        stat[bg * 2 + 1] = rsqrtf(var + 1e-5f);
    }
}

__global__ void gn_apply_kernel(float* __restrict__ v, const float* __restrict__ stat,
                                const float* __restrict__ g, const float* __restrict__ gb,
                                int HW, int total) {
    int idx = blockIdx.x * 256 + threadIdx.x;
    if (idx >= total) return;
    int c = (idx / HW) & 255;
    int b = idx / (HW * 256);
    int grp = c >> 4;
    float m = stat[(b * 16 + grp) * 2];
    float r = stat[(b * 16 + grp) * 2 + 1];
    v[idx] = (v[idx] - m) * r * g[c] + gb[c];
}

// ---------------------------------------------------------------------------
// Per-(b,c) spatial mean.
// ---------------------------------------------------------------------------
__global__ void chanmean_kernel(const float* __restrict__ v, int HW, float* __restrict__ cm) {
    __shared__ float ss[256];
    int bc = blockIdx.x;
    const float* base = v + (size_t)bc * HW;
    float s = 0.0f;
    for (int i = threadIdx.x; i < HW; i += 256) s += base[i];
    ss[threadIdx.x] = s;
    __syncthreads();
    for (int st = 128; st > 0; st >>= 1) {
        if (threadIdx.x < st) ss[threadIdx.x] += ss[threadIdx.x + st];
        __syncthreads();
    }
    if (threadIdx.x == 0) cm[bc] = ss[0] / HW;
}

__global__ void attn_kernel(const float* __restrict__ cm, const float* __restrict__ aw,
                            const float* __restrict__ ab, float* __restrict__ s) {
    __shared__ float ss[256];
    int b = blockIdx.x;
    ss[threadIdx.x] = cm[b * 256 + threadIdx.x] * aw[threadIdx.x];
    __syncthreads();
    for (int st = 128; st > 0; st >>= 1) {
        if (threadIdx.x < st) ss[threadIdx.x] += ss[threadIdx.x + st];
        __syncthreads();
    }
    if (threadIdx.x == 0) s[b] = hsig(fmaxf(ss[0] + ab[0], 0.0f));
}

// ---------------------------------------------------------------------------
// Bilinear upsample (linspace endpoints, floor clipped to [0, h-2]).
// ---------------------------------------------------------------------------
__global__ void upsample_kernel(const float* __restrict__ in, int h, int w,
                                float* __restrict__ out, int H, int W, int total) {
    int idx = blockIdx.x * 256 + threadIdx.x;
    if (idx >= total) return;
    int HW = H * W;
    int bc = idx / HW;
    int sp = idx - bc * HW;
    int y = sp / W, xx = sp - y * W;
    float ys = y  * (float)(h - 1) / (float)(H - 1);
    float xs = xx * (float)(w - 1) / (float)(W - 1);
    int y0 = min(max((int)floorf(ys), 0), h - 2);
    int x0 = min(max((int)floorf(xs), 0), w - 2);
    float wy = ys - y0, wxv = xs - x0;
    const float* ib = in + (size_t)bc * h * w;
    out[idx] = ib[y0 * w + x0] * (1 - wy) * (1 - wxv)
             + ib[y0 * w + x0 + 1] * (1 - wy) * wxv
             + ib[(y0 + 1) * w + x0] * wy * (1 - wxv)
             + ib[(y0 + 1) * w + x0 + 1] * wy * wxv;
}

__global__ void combine_kernel(const float* __restrict__ f0, const float* __restrict__ f1,
                               const float* __restrict__ f2, const float* __restrict__ s,
                               int nf, int CHW, float* __restrict__ out, int total) {
    int idx = blockIdx.x * 256 + threadIdx.x;
    if (idx >= total) return;
    int b = idx / CHW;
    float acc = s[b] * f0[idx];
    if (nf > 1) acc += s[4 + b] * f1[idx];
    if (nf > 2) acc += s[8 + b] * f2[idx];
    out[idx] = acc / (float)nf;
}

// ---------------------------------------------------------------------------
// DyReLU: FC(256->64)+relu, FC(64->1024)+hard_sigmoid, then elementwise max.
// ---------------------------------------------------------------------------
__global__ void dyrelu_coef_kernel(const float* __restrict__ p, const float* __restrict__ w1,
                                   const float* __restrict__ b1, const float* __restrict__ w2,
                                   const float* __restrict__ b2, float* __restrict__ coef) {
    __shared__ float h[64];
    int b = blockIdx.x, tid = threadIdx.x;
    if (tid < 64) {
        float a = b1[tid];
        const float* wr = w1 + tid * 256;
        const float* pb = p + b * 256;
        for (int c = 0; c < 256; ++c) a += wr[c] * pb[c];
        h[tid] = fmaxf(a, 0.0f);
    }
    __syncthreads();
    for (int o = tid; o < 1024; o += 256) {
        float a = b2[o];
        const float* wr = w2 + o * 64;
#pragma unroll 8
        for (int j = 0; j < 64; ++j) a += wr[j] * h[j];
        coef[b * 1024 + o] = hsig(a);
    }
}

__global__ void dyrelu_apply_kernel(const float* __restrict__ v, const float* __restrict__ coef,
                                    float* __restrict__ out, int HW, int total) {
    int idx = blockIdx.x * 256 + threadIdx.x;
    if (idx >= total) return;
    int c = (idx / HW) & 255;
    int b = idx / (HW * 256);
    const float* cf = coef + b * 1024;
    float a1  = (cf[c] - 0.5f) * 2.0f + 1.0f;
    float bb1 =  cf[256 + c] - 0.5f;
    float a2  = (cf[512 + c] - 0.5f) * 2.0f;
    float bb2 =  cf[768 + c] - 0.5f;
    float xv = v[idx];
    out[idx] = fmaxf(xv * a1 + bb1, xv * a2 + bb2);
}

// ---------------------------------------------------------------------------
extern "C" void kernel_launch(void* const* d_in, const int* in_sizes, int n_in,
                              void* d_out, int out_size, void* d_ws, size_t ws_size,
                              hipStream_t stream) {
    const float* x0     = (const float*)d_in[0];
    const float* x1     = (const float*)d_in[1];
    const float* x2     = (const float*)d_in[2];
    const float* off_w  = (const float*)d_in[3];
    const float* off_b  = (const float*)d_in[4];
    const float* w0     = (const float*)d_in[5];
    const float* b0     = (const float*)d_in[6];
    const float* g0     = (const float*)d_in[7];
    const float* gb0    = (const float*)d_in[8];
    const float* w1     = (const float*)d_in[9];
    const float* b1     = (const float*)d_in[10];
    const float* g1     = (const float*)d_in[11];
    const float* gb1    = (const float*)d_in[12];
    const float* w2     = (const float*)d_in[13];
    const float* b2     = (const float*)d_in[14];
    const float* g2     = (const float*)d_in[15];
    const float* gb2    = (const float*)d_in[16];
    const float* attn_w = (const float*)d_in[17];
    const float* attn_b = (const float*)d_in[18];
    const float* fc1_w  = (const float*)d_in[19];
    const float* fc1_b  = (const float*)d_in[20];
    const float* fc2_w  = (const float*)d_in[21];
    const float* fc2_b  = (const float*)d_in[22];

    const int B = 4, C = 256;

    // Workspace carve (byte bump allocator, 256B aligned).
    char* ws = (char*)d_ws;
    size_t off = 0;
    auto alloc = [&](size_t bytes) -> void* {
        void* p = ws + off;
        off += (bytes + 255) & ~(size_t)255;
        return p;
    };
    unsigned short* wf0 = (unsigned short*)alloc(589824 * 2);
    unsigned short* wf1 = (unsigned short*)alloc(589824 * 2);
    unsigned short* wf2 = (unsigned short*)alloc(589824 * 2);
    float* om      = (float*)alloc((size_t)B * 27 * 6400 * 4);
    float* fea[3];
    fea[0] = (float*)alloc((size_t)B * C * 6400 * 4);
    fea[1] = (float*)alloc((size_t)B * C * 6400 * 4);
    fea[2] = (float*)alloc((size_t)B * C * 6400 * 4);
    float* lobuf   = (float*)alloc((size_t)B * C * 1600 * 4);
    float* meanfea = (float*)alloc((size_t)B * C * 6400 * 4);
    float* gnstat  = (float*)alloc((size_t)B * 16 * 2 * 4);
    float* cmean   = (float*)alloc((size_t)B * C * 4);
    float* attns   = (float*)alloc(16 * 4);
    float* pbuf    = (float*)alloc((size_t)B * C * 4);
    float* coef    = (float*)alloc((size_t)B * 1024 * 4);

    // Weight pre-shuffle to bf16 WMMA fragments (once per call).
    wfrag_kernel<<<2304, 256, 0, stream>>>(w0, wf0);
    wfrag_kernel<<<2304, 256, 0, stream>>>(w1, wf1);
    wfrag_kernel<<<2304, 256, 0, stream>>>(w2, wf2);

    auto run_level = [&](const float* feat, int H, int W,
                         const float* prev, int pH, int pW,
                         const float* nxt, int nH, int nW,
                         float* outp) {
        const int HW = H * W;
        const int total = B * C * HW;
        const int eb = (total + 255) / 256;

        offconv_kernel<<<(B * 27 * HW + 255) / 256, 256, 0, stream>>>(feat, off_w, off_b, om, H, W);

        // fea[0]: self branch (w1/g1, stride 1)
        deform_wmma_kernel<<<B * HW / 16, 256, 0, stream>>>(feat, H, W, om, H, W, H, W, 1, wf1, b1, fea[0]);
        gn_stats_kernel<<<B * 16, 256, 0, stream>>>(fea[0], HW, gnstat);
        gn_apply_kernel<<<eb, 256, 0, stream>>>(fea[0], gnstat, g1, gb1, HW, total);
        chanmean_kernel<<<B * C, 256, 0, stream>>>(fea[0], HW, cmean);
        attn_kernel<<<B, 256, 0, stream>>>(cmean, attn_w, attn_b, attns + 0);

        int slot = 1;
        if (prev) {   // stride-2 branch (w2/g2)
            deform_wmma_kernel<<<B * HW / 16, 256, 0, stream>>>(prev, pH, pW, om, H, W, H, W, 2, wf2, b2, fea[slot]);
            gn_stats_kernel<<<B * 16, 256, 0, stream>>>(fea[slot], HW, gnstat);
            gn_apply_kernel<<<eb, 256, 0, stream>>>(fea[slot], gnstat, g2, gb2, HW, total);
            chanmean_kernel<<<B * C, 256, 0, stream>>>(fea[slot], HW, cmean);
            attn_kernel<<<B, 256, 0, stream>>>(cmean, attn_w, attn_b, attns + slot * 4);
            slot++;
        }
        if (nxt) {    // low-res branch (w0/g0) + bilinear upsample
            const int ho = nH, wo = nW, hw2 = ho * wo;
            const int tot2 = B * C * hw2;
            deform_wmma_kernel<<<B * hw2 / 16, 256, 0, stream>>>(nxt, nH, nW, om, H, W, ho, wo, 1, wf0, b0, lobuf);
            gn_stats_kernel<<<B * 16, 256, 0, stream>>>(lobuf, hw2, gnstat);
            gn_apply_kernel<<<(tot2 + 255) / 256, 256, 0, stream>>>(lobuf, gnstat, g0, gb0, hw2, tot2);
            upsample_kernel<<<eb, 256, 0, stream>>>(lobuf, ho, wo, fea[slot], H, W, total);
            chanmean_kernel<<<B * C, 256, 0, stream>>>(fea[slot], HW, cmean);
            attn_kernel<<<B, 256, 0, stream>>>(cmean, attn_w, attn_b, attns + slot * 4);
            slot++;
        }

        combine_kernel<<<eb, 256, 0, stream>>>(fea[0], fea[1], (slot > 2) ? fea[2] : nullptr,
                                               attns, slot, C * HW, meanfea, total);

        chanmean_kernel<<<B * C, 256, 0, stream>>>(meanfea, HW, pbuf);
        dyrelu_coef_kernel<<<B, 256, 0, stream>>>(pbuf, fc1_w, fc1_b, fc2_w, fc2_b, coef);
        dyrelu_apply_kernel<<<eb, 256, 0, stream>>>(meanfea, coef, outp, HW, total);
    };

    float* outF = (float*)d_out;
    run_level(x0, 80, 80, nullptr, 0, 0, x1, 40, 40, outF);
    run_level(x1, 40, 40, x0, 80, 80, x2, 20, 20, outF + (size_t)B * C * 6400);
    run_level(x2, 20, 20, x1, 40, 40, nullptr, 0, 0, outF + (size_t)B * C * (6400 + 1600));
}
